// GraphSAGEEncoder_64192581206427
// MI455X (gfx1250) — compile-verified, matrix operand order checked by reading, not run on hardware
//
#include <hip/hip_runtime.h>

// ---------------------------------------------------------------------------
// GraphSAGE encoder for MI455X (gfx1250): 3x (SAGEConv + BN [+ReLU])
//   N=50000 nodes, E=800000 edges, 128 -> 256 -> 256 -> 256
// GEMMs on v_wmma_f32_16x16x32_bf16 (fp32 accumulate); aggregation is an
// L2-resident float4 atomic scatter (feature table fits in the 192MB L2).
// Weight tiles staged LDS-async (GLOBAL_LOAD_ASYNC_TO_LDS_B128) when the
// toolchain exposes the builtin; sync fallback otherwise.
// ---------------------------------------------------------------------------

#define NN 50000
#define EE 800000
#define EPS 1e-5f

typedef __attribute__((ext_vector_type(16))) __bf16 v16bf;
typedef __attribute__((ext_vector_type(8)))  float  v8f;
typedef __attribute__((ext_vector_type(4)))  int    v4i;

union V16 { v16bf bf; unsigned short u[16]; };

#if defined(__has_builtin)
#if __has_builtin(__builtin_amdgcn_global_load_async_to_lds_b128) && \
    __has_builtin(__builtin_amdgcn_s_wait_asynccnt)
#define HAVE_ASYNC_LDS 1
#endif
#endif

__device__ __forceinline__ unsigned short f2bf(float f) {
    unsigned int u = __float_as_uint(f);
    u = (u + 0x7FFFu + ((u >> 16) & 1u)) >> 16;   // round-to-nearest-even
    return (unsigned short)u;
}

// ------------------------------- utility -----------------------------------
__global__ __launch_bounds__(256) void zero_f32(float* __restrict__ p, int n) {
    int i = blockIdx.x * 256 + threadIdx.x;
    if (i < n) p[i] = 0.0f;
}

__global__ __launch_bounds__(256) void deg_accum(const int* __restrict__ dst,
                                                 float* __restrict__ deg, int e) {
    int i = blockIdx.x * 256 + threadIdx.x;
    if (i < e) atomicAdd(&deg[dst[i]], 1.0f);
}

__global__ __launch_bounds__(256) void inv_deg(const float* __restrict__ deg,
                                               float* __restrict__ inv, int n) {
    int i = blockIdx.x * 256 + threadIdx.x;
    if (i < n) inv[i] = 1.0f / fmaxf(deg[i], 1.0f);
}

// Pack [Wl | Wr] (row co: Wl[co][0..Ci-1], Wr[co][0..Ci-1]) as bf16, K = 2*Ci.
__global__ __launch_bounds__(256) void cvt_weights(const float* __restrict__ Wl,
                                                   const float* __restrict__ Wr,
                                                   unsigned short* __restrict__ Wcat,
                                                   int Ci, int total /* Co*2*Ci */) {
    int i = blockIdx.x * 256 + threadIdx.x;
    if (i >= total) return;
    int K  = 2 * Ci;
    int co = i / K;
    int k  = i - co * K;
    float v = (k < Ci) ? Wl[co * Ci + k] : Wr[co * Ci + (k - Ci)];
    Wcat[i] = f2bf(v);
}

// ------------------------- edge scatter (segment sum) ----------------------
__global__ __launch_bounds__(256) void scatter_add(const float* __restrict__ h,
                                                   const int* __restrict__ src,
                                                   const int* __restrict__ dst,
                                                   float* __restrict__ agg,
                                                   int C4, int total /* E*C4 */) {
    int idx = blockIdx.x * 256 + threadIdx.x;
    if (idx >= total) return;
    int e = idx / C4;
    int c = idx - e * C4;
    int C = C4 * 4;
    // hide L2 latency of the random gather: prefetch a future edge's row
    if (c == 0 && idx + 16 * C4 < total)
        __builtin_prefetch(h + (size_t)src[e + 16] * C, 0, 0);
    const float4 v = ((const float4*)(h + (size_t)src[e] * C))[c];
    float* o = agg + (size_t)dst[e] * C + c * 4;
    atomicAdd(o + 0, v.x);
    atomicAdd(o + 1, v.y);
    atomicAdd(o + 2, v.z);
    atomicAdd(o + 3, v.w);
}

// ----------------------------- WMMA GEMM -----------------------------------
// out[n, co] = sum_k  [mean(n,:) | h(n,:)][k] * Wcat[co][k]  + bias[co]
// mean(n,k) = agg[n,k] * invdeg[n] applied at tile-load time.
// Block: 256 thr = 8 waves; tile M=32 x N=128; each wave owns a 16x32 tile
// (1 A fragment, 2 B fragments, 2 WMMAs per k-step).
#define MBLK 32
#define NBLK 128
#define KBLK 32
#define KPAD 8                       // +16B row pad -> conflict-free b128 reads
#define KROW (KBLK + KPAD)           // 40 ushorts = 80B

template <int CI>
__global__ __launch_bounds__(256)
void gemm_sage_wmma(const float* __restrict__ agg,
                    const float* __restrict__ invdeg,
                    const float* __restrict__ h,
                    const unsigned short* __restrict__ Wb,  // [Co][K] bf16
                    const float* __restrict__ bias,
                    float* __restrict__ out,
                    int Nn, int Co) {
    constexpr int K = 2 * CI;
    __shared__ unsigned short As[MBLK][KROW];   // 2.5 KB
    __shared__ unsigned short Bs[NBLK][KROW];   // 10 KB

    const int m0   = blockIdx.x * MBLK;
    const int n0   = blockIdx.y * NBLK;
    const int tid  = threadIdx.x;
    const int wave = tid >> 5;        // 0..7
    const int lane = tid & 31;
    const int wm   = wave >> 2;       // 0..1 (row tile)
    const int wn   = wave & 3;        // 0..3 (32-wide col tile)
    const int half = lane >> 4;
    const int l15  = lane & 15;

    // A-staging thread mapping (hoisted)
    const int ar  = tid >> 3;               // 0..31 row
    const int akq = (tid & 7) * 4;          // 0..28
    const int an  = m0 + ar;
    const bool avalid = (an < Nn);
    const float ainv = avalid ? invdeg[an] : 0.0f;
    const float* __restrict__ aggrow = agg + (size_t)(avalid ? an : 0) * CI + akq;
    const float* __restrict__ hrow   = h   + (size_t)(avalid ? an : 0) * CI + akq;

    // B-staging thread mapping: 128 rows x 32 k = 4096 elems, 16 per thread
    const int br  = tid >> 1;               // 0..127
    const int bkq = (tid & 1) * 16;         // 0 or 16
    const unsigned short* __restrict__ wrow = Wb + (size_t)(n0 + br) * K + bkq;

    v8f c0 = {};
    v8f c1 = {};

    for (int kk = 0; kk < K; kk += KBLK) {
        // ---- stage A tile (32 x 32), fp32 -> bf16, mean fused ----
        {
            float4 v = {0.f, 0.f, 0.f, 0.f};
            float  s = 1.0f;
            if (avalid) {
                if (kk + akq < CI) {          // uniform per k-step (KBLK | CI)
                    v = *(const float4*)(aggrow + kk);
                    s = ainv;
                } else {
                    v = *(const float4*)(hrow + (kk - CI));
                }
            }
            As[ar][akq + 0] = f2bf(v.x * s);
            As[ar][akq + 1] = f2bf(v.y * s);
            As[ar][akq + 2] = f2bf(v.z * s);
            As[ar][akq + 3] = f2bf(v.w * s);
        }
        // ---- stage B tile (128 x 32), already bf16: 2 x 16B per thread ----
#if HAVE_ASYNC_LDS
        __builtin_amdgcn_global_load_async_to_lds_b128(
            (__attribute__((address_space(1))) v4i*)(wrow + kk),
            (__attribute__((address_space(3))) v4i*)&Bs[br][bkq], 0, 0);
        __builtin_amdgcn_global_load_async_to_lds_b128(
            (__attribute__((address_space(1))) v4i*)(wrow + kk + 8),
            (__attribute__((address_space(3))) v4i*)&Bs[br][bkq + 8], 0, 0);
        __builtin_amdgcn_s_wait_asynccnt(0);
#else
        *(uint4*)&Bs[br][bkq]     = *(const uint4*)(wrow + kk);
        *(uint4*)&Bs[br][bkq + 8] = *(const uint4*)(wrow + kk + 8);
#endif
        __syncthreads();

        // ---- build fragments per ISA 16-bit A 16x32 layout ----
        V16 a, b0, b1;
        const int arow  = wm * 16 + l15;
        const int brow0 = wn * 32 + l15;
        const int brow1 = brow0 + 16;
        const int kb    = half * 8;
#pragma unroll
        for (int e = 0; e < 8; ++e) {
            a.u[e]      = As[arow][kb + e];
            a.u[e + 8]  = As[arow][16 + kb + e];
            b0.u[e]     = Bs[brow0][kb + e];
            b0.u[e + 8] = Bs[brow0][16 + kb + e];
            b1.u[e]     = Bs[brow1][kb + e];
            b1.u[e + 8] = Bs[brow1][16 + kb + e];
        }
        c0 = __builtin_amdgcn_wmma_f32_16x16x32_bf16(false, a.bf, false, b0.bf,
                                                     (short)0, c0, false, false);
        c1 = __builtin_amdgcn_wmma_f32_16x16x32_bf16(false, a.bf, false, b1.bf,
                                                     (short)0, c1, false, false);
        __syncthreads();
    }

    // ---- store D (C/D layout: vgpr r -> M = r + half*8, N = l15) + bias ----
    const int co0 = n0 + wn * 32 + l15;
    const int co1 = co0 + 16;
    const float bv0 = bias[co0];
    const float bv1 = bias[co1];
#pragma unroll
    for (int r = 0; r < 8; ++r) {
        int node = m0 + wm * 16 + half * 8 + r;
        if (node < Nn) {
            out[(size_t)node * Co + co0] = c0[r] + bv0;
            out[(size_t)node * Co + co1] = c1[r] + bv1;
        }
    }
}

// ----------------------------- BatchNorm -----------------------------------
// Per-channel sums: 1 thread per channel, 256 rows per block, 1 atomic/chan.
__global__ __launch_bounds__(256) void bn_stats(const float* __restrict__ x,
                                                float* __restrict__ sum,
                                                float* __restrict__ sumsq,
                                                int Nn, int C) {
    int c  = threadIdx.x;               // C == 256 == blockDim.x
    int r0 = blockIdx.x * 256;
    int r1 = min(r0 + 256, Nn);
    float s = 0.f, q = 0.f;
    for (int r = r0; r < r1; ++r) {
        float v = x[(size_t)r * C + c];
        s += v;
        q += v * v;
    }
    atomicAdd(&sum[c], s);
    atomicAdd(&sumsq[c], q);
}

__global__ __launch_bounds__(256) void bn_finalize(const float* __restrict__ sum,
                                                   const float* __restrict__ sumsq,
                                                   const float* __restrict__ g,
                                                   const float* __restrict__ b,
                                                   float* __restrict__ scale,
                                                   float* __restrict__ shift,
                                                   int Nn) {
    int c = threadIdx.x;
    float invN = 1.0f / (float)Nn;
    float mu   = sum[c] * invN;
    float var  = sumsq[c] * invN - mu * mu;
    float sc   = g[c] * rsqrtf(var + EPS);
    scale[c] = sc;
    shift[c] = b[c] - mu * sc;
}

__global__ __launch_bounds__(256) void bn_apply(const float* __restrict__ x,
                                                const float* __restrict__ scale,
                                                const float* __restrict__ shift,
                                                float* __restrict__ y,
                                                int total, int C, int do_relu) {
    int i = blockIdx.x * 256 + threadIdx.x;
    if (i >= total) return;
    int c = i % C;
    float v = fmaf(x[i], scale[c], shift[c]);
    y[i] = do_relu ? fmaxf(v, 0.0f) : v;
}

// ----------------------------- orchestration -------------------------------
extern "C" void kernel_launch(void* const* d_in, const int* in_sizes, int n_in,
                              void* d_out, int out_size, void* d_ws, size_t ws_size,
                              hipStream_t stream) {
    const int Nn = NN, E = EE;
    const float* x   = (const float*)d_in[0];
    const int*   ei  = (const int*)d_in[1];
    const int*   src = ei;
    const int*   dst = ei + E;

    struct Layer { const float *Wl, *bl, *Wr, *g, *b; int Ci, Co; };
    Layer L[3] = {
        { (const float*)d_in[2],  (const float*)d_in[3],  (const float*)d_in[4],
          (const float*)d_in[5],  (const float*)d_in[6],  128, 256 },
        { (const float*)d_in[7],  (const float*)d_in[8],  (const float*)d_in[9],
          (const float*)d_in[10], (const float*)d_in[11], 256, 256 },
        { (const float*)d_in[12], (const float*)d_in[13], (const float*)d_in[14],
          (const float*)d_in[15], (const float*)d_in[16], 256, 256 },
    };

    // ---- carve workspace (aligned 256B) ----
    char*  wsp = (char*)d_ws;
    size_t off = 0;
    auto take = [&](size_t bytes) -> void* {
        void* p = wsp + off;
        off += (bytes + 255) & ~(size_t)255;
        return p;
    };
    float*          h_cur  = (float*)take((size_t)Nn * 256 * 4);
    float*          agg    = (float*)take((size_t)Nn * 256 * 4);
    float*          deg    = (float*)take((size_t)Nn * 4);
    float*          invdeg = (float*)take((size_t)Nn * 4);
    unsigned short* Wcat   = (unsigned short*)take((size_t)256 * 512 * 2);
    float*          bnsum  = (float*)take(256 * 4);
    float*          bnsq   = (float*)take(256 * 4);
    float*          bnsc   = (float*)take(256 * 4);
    float*          bnsh   = (float*)take(256 * 4);
    (void)ws_size; (void)in_sizes; (void)n_in; (void)out_size;

    // ---- degrees (shared by all layers) ----
    zero_f32 <<<(Nn + 255) / 256, 256, 0, stream>>>(deg, Nn);
    deg_accum<<<(E + 255) / 256, 256, 0, stream>>>(dst, deg, E);
    inv_deg  <<<(Nn + 255) / 256, 256, 0, stream>>>(deg, invdeg, Nn);

    const float* h   = x;              // layer-0 input (Ci = 128)
    float*       out = (float*)d_out;  // per-layer GEMM output buffer

    for (int i = 0; i < 3; ++i) {
        const int Ci = L[i].Ci, Co = L[i].Co, K = 2 * Ci;

        // weights -> bf16 concat [Wl | Wr]
        int wtot = Co * K;
        cvt_weights<<<(wtot + 255) / 256, 256, 0, stream>>>(L[i].Wl, L[i].Wr, Wcat, Ci, wtot);

        // mean aggregation: zero + scatter
        int atot = Nn * Ci;
        zero_f32<<<(atot + 255) / 256, 256, 0, stream>>>(agg, atot);
        int C4 = Ci / 4;
        int stot = E * C4;
        scatter_add<<<(stot + 255) / 256, 256, 0, stream>>>(h, src, dst, agg, C4, stot);

        // fused [mean|h] @ [Wl|Wr]^T + bias on WMMA
        dim3 gg((Nn + MBLK - 1) / MBLK, Co / NBLK);
        if (Ci == 128)
            gemm_sage_wmma<128><<<gg, 256, 0, stream>>>(agg, invdeg, h, Wcat, L[i].bl, out, Nn, Co);
        else
            gemm_sage_wmma<256><<<gg, 256, 0, stream>>>(agg, invdeg, h, Wcat, L[i].bl, out, Nn, Co);

        // batchnorm (+ relu for layers 0,1)
        zero_f32<<<1, 256, 0, stream>>>(bnsum, 256);
        zero_f32<<<1, 256, 0, stream>>>(bnsq, 256);
        bn_stats<<<(Nn + 255) / 256, 256, 0, stream>>>(out, bnsum, bnsq, Nn, Co);
        bn_finalize<<<1, 256, 0, stream>>>(bnsum, bnsq, L[i].g, L[i].b, bnsc, bnsh, Nn);
        float* dsty = (i < 2) ? h_cur : out;   // last layer normalizes in place
        int ntot = Nn * Co;
        bn_apply<<<(ntot + 255) / 256, 256, 0, stream>>>(out, bnsc, bnsh, dsty, ntot, Co, i < 2 ? 1 : 0);

        h = h_cur;
    }
}
